// Model_v2_20186346291366
// MI455X (gfx1250) — matrix operand order repaired and checked by reading, not compiled, not run on hardware
//
#include <hip/hip_runtime.h>

// ---------------- constants ----------------
#define P_TOT 28800      // B*H*W = 128*15*15
#define NB    128        // batch
#define HW    225
#define MAPC  256
#define GC    64

typedef unsigned short u16;
typedef __attribute__((ext_vector_type(16))) __bf16 v16bf;
typedef __attribute__((ext_vector_type(8)))  float  v8f;

union FragAB { v16bf v; uint4 q[2]; };
union Pack8  { uint4 q; u16 s[8]; };

__constant__ int c_DR[4][3] = {{0,0,0},{-1,0,1},{-1,0,1},{1,0,-1}};
__constant__ int c_DC[4][3] = {{-1,0,1},{0,0,0},{-1,0,1},{-1,0,1}};

__device__ __forceinline__ u16 f2bf(float f) {
  unsigned int u = __float_as_uint(f);
  unsigned int r = (u + 0x7FFFu + ((u >> 16) & 1u)) >> 16;
  return (u16)r;
}
__device__ __forceinline__ float bf2f(u16 b) {
  return __uint_as_float(((unsigned int)b) << 16);
}
__device__ __forceinline__ float swishf(float x) {
  return x / (1.f + __expf(-x));
}
__device__ __forceinline__ float bslope(float s, float bound) {
  return tanhf(s / bound) * bound;
}

// ---------------- weight prep ----------------
__global__ void k_f32_to_bf16(u16* __restrict__ dst, const float* __restrict__ src, int n) {
  int i = blockIdx.x * 256 + threadIdx.x;
  if (i < n) dst[i] = f2bf(src[i]);
}

// rb_w [4][3][256][256] -> Wline [4][256][768]  (k = t*256 + ic)
__global__ void k_line_weights(u16* __restrict__ dst, const float* __restrict__ src) {
  int i = blockIdx.x * 256 + threadIdx.x;
  const int n = 4 * 3 * 256 * 256;
  if (i >= n) return;
  int ic = i & 255;
  int oc = (i >> 8) & 255;
  int jt = i >> 16;          // j*3 + t
  int t  = jt % 3;
  int j  = jt / 3;
  dst[((size_t)(j * 256 + oc)) * 768 + t * 256 + ic] = f2bf(src[i]);
}

// ---------------- initial line conv (K=6, VALU) ----------------
// y[d][p][c] = swish(line3(xin, fw, d) + fb)
__global__ void k_initial(const float* __restrict__ x, const float* __restrict__ fw,
                          const float* __restrict__ fb, u16* __restrict__ y) {
  int p = blockIdx.x;
  int c = threadIdx.x;
  int b = p / HW, hw = p - b * HW;
  int h = hw / 15, w = hw - h * 15;
  const float* xb = x + (size_t)b * 2 * HW;
  float bias = fb[c];
  for (int d = 0; d < 4; ++d) {
    float val = bias;
    #pragma unroll
    for (int t = 0; t < 3; ++t) {
      int nh = h + c_DR[d][t], nw = w + c_DC[d][t];
      if ((unsigned)nh < 15u && (unsigned)nw < 15u) {
        int o = nh * 15 + nw;
        float x0 = xb[o], x1 = xb[HW + o];
        const float* wv = fw + ((size_t)t * 256 + c) * 2;
        val += wv[0] * x0 + wv[1] * x1;
      }
    }
    y[(size_t)d * P_TOT * 256 + (size_t)p * 256 + c] = f2bf(swishf(val));
  }
}

// ---------------- WMMA GEMM ----------------
// Out[p][m] = epi( sum_k A[m][k] * Bact(k,p) + bias[m] )
// BMODE 0: Bact(k,p) = B[p][k]          (K = M-channels of B)
// BMODE 1: Bact(k,p) = B[shift(p,d,t)][c], k = t*256 + c  (3-tap line, K=768)
// EPI 0: swish -> bf16 ; EPI 1: swish + Res -> bf16 ; EPI 2: 30*tanh(x/30) -> f32
template<int EPI, int BMODE>
__global__ __launch_bounds__(256, 1)
void k_gemm(const u16* __restrict__ A, const u16* __restrict__ B,
            const float* __restrict__ bias, void* __restrict__ OutP,
            const u16* __restrict__ Res, int M, int K, int dir) {
  __shared__ u16 As[2][128][32];
  __shared__ u16 Bs[2][128][32];

  const int tid  = threadIdx.x;
  const int lane = tid & 31;
  const int wave = tid >> 5;
  const int wm   = wave >> 1;     // 0..3 -> 32 rows of M
  const int wn   = wave & 1;      // 0..1 -> 64 cols of N
  const int row  = tid >> 1;      // 0..127
  const int hf   = tid & 1;       // half-row (16 elems)
  const int n0   = blockIdx.x * 128;
  const int m0   = blockIdx.y * 128;

  // geometry of this thread's B row (position) for line mode
  const int pRow  = n0 + row;
  const int hwRow = pRow % HW;
  const int hRow  = hwRow / 15;
  const int wRow  = hwRow - hRow * 15;

  const int nK = K >> 5;

  v8f acc[2][4];
  #pragma unroll
  for (int i = 0; i < 2; ++i)
    #pragma unroll
    for (int j = 0; j < 4; ++j)
      #pragma unroll
      for (int r = 0; r < 8; ++r) acc[i][j][r] = 0.f;

  uint4 ra0, ra1, rb0, rb1;
  auto loadG = [&](int kb) {
    const u16* ap = A + (size_t)(m0 + row) * K + kb * 32 + hf * 16;
    ra0 = *(const uint4*)ap;
    ra1 = *(const uint4*)(ap + 8);
    if (BMODE == 0) {
      const u16* bp = B + (size_t)pRow * K + kb * 32 + hf * 16;
      rb0 = *(const uint4*)bp;
      rb1 = *(const uint4*)(bp + 8);
    } else {
      // branchless shifted load: clamp to a safe address, mask result to zero
      int k0 = kb * 32;
      int t  = k0 >> 8;               // tap 0..2 (uniform)
      int di = t - 1;                 // -1,0,1 (uniform)
      int dr = (dir == 0) ? 0 : ((dir == 3) ? -di : di);   // uniform
      int dc = (dir == 1) ? 0 : di;                        // uniform
      int nh = hRow + dr, nw = wRow + dc;
      bool ok = ((unsigned)nh < 15u) & ((unsigned)nw < 15u);
      int srcP = ok ? (pRow + dr * 15 + dc) : pRow;        // v_cndmask
      unsigned msk = ok ? 0xFFFFFFFFu : 0u;                // v_cndmask
      const u16* bp = B + (size_t)srcP * 256 + (k0 & 255) + hf * 16;
      uint4 t0 = *(const uint4*)bp;
      uint4 t1 = *(const uint4*)(bp + 8);
      rb0.x = t0.x & msk; rb0.y = t0.y & msk; rb0.z = t0.z & msk; rb0.w = t0.w & msk;
      rb1.x = t1.x & msk; rb1.y = t1.y & msk; rb1.z = t1.z & msk; rb1.w = t1.w & msk;
    }
  };

  loadG(0);
  int buf = 0;
  const int g  = lane >> 4;
  const int lr = lane & 15;
  for (int kb = 0; kb < nK; ++kb) {
    *(uint4*)&As[buf][row][hf * 16]     = ra0;
    *(uint4*)&As[buf][row][hf * 16 + 8] = ra1;
    *(uint4*)&Bs[buf][row][hf * 16]     = rb0;
    *(uint4*)&Bs[buf][row][hf * 16 + 8] = rb1;
    __syncthreads();
    if (kb + 1 < nK) loadG(kb + 1);

    FragAB fa[2], fbb[4];
    #pragma unroll
    for (int mt = 0; mt < 2; ++mt) {
      int m = wm * 32 + mt * 16 + lr;
      fa[mt].q[0] = *(const uint4*)&As[buf][m][g * 8];
      fa[mt].q[1] = *(const uint4*)&As[buf][m][16 + g * 8];
    }
    #pragma unroll
    for (int nt = 0; nt < 4; ++nt) {
      int n = wn * 64 + nt * 16 + lr;
      fbb[nt].q[0] = *(const uint4*)&Bs[buf][n][g * 8];
      fbb[nt].q[1] = *(const uint4*)&Bs[buf][n][16 + g * 8];
    }
    #pragma unroll
    for (int mt = 0; mt < 2; ++mt)
      #pragma unroll
      for (int nt = 0; nt < 4; ++nt)
        acc[mt][nt] = __builtin_amdgcn_wmma_f32_16x16x32_bf16(
            false, fa[mt].v, false, fbb[nt].v, (short)0, acc[mt][nt], false, false);
    __syncthreads();
    buf ^= 1;
  }

  // epilogue: lane holds m = mg..mg+7 (consecutive), n = ng
  #pragma unroll
  for (int mt = 0; mt < 2; ++mt) {
    #pragma unroll
    for (int nt = 0; nt < 4; ++nt) {
      int mg = m0 + wm * 32 + mt * 16 + g * 8;
      int ng = n0 + wn * 64 + nt * 16 + lr;
      float v[8];
      #pragma unroll
      for (int r = 0; r < 8; ++r) {
        float x = acc[mt][nt][r] + bias[mg + r];
        if (EPI == 2) v[r] = 30.f * tanhf(x * (1.f / 30.f));
        else          v[r] = swishf(x);
      }
      if (EPI == 1) {
        Pack8 rp;
        rp.q = *(const uint4*)(Res + (size_t)ng * M + mg);
        #pragma unroll
        for (int r = 0; r < 8; ++r) v[r] += bf2f(rp.s[r]);
      }
      if (EPI == 2) {
        float* Of = (float*)OutP + (size_t)ng * M + mg;
        *(float4*)(Of)     = make_float4(v[0], v[1], v[2], v[3]);
        *(float4*)(Of + 4) = make_float4(v[4], v[5], v[6], v[7]);
      } else {
        Pack8 op;
        #pragma unroll
        for (int r = 0; r < 8; ++r) op.s[r] = f2bf(v[r]);
        *(uint4*)((u16*)OutP + (size_t)ng * M + mg) = op.q;
      }
    }
  }
}

// ---------------- head (fp32) ----------------
// h1[p][c] = prelu(mean_d mapf[d][p][c], g1lr, bound .999)   (c < 64)
__global__ void k_h1(const float* __restrict__ mapf, const float* __restrict__ g1lr,
                     float* __restrict__ h1) {
  int i = blockIdx.x * 256 + threadIdx.x;
  if (i >= P_TOT * GC) return;
  int c = i & 63, p = i >> 6;
  size_t o = (size_t)p * 128 + c;
  float m = 0.25f * (mapf[o] + mapf[(size_t)P_TOT * 128 + o] +
                     mapf[2 * (size_t)P_TOT * 128 + o] + mapf[3 * (size_t)P_TOT * 128 + o]);
  float s = bslope(g1lr[c], 0.999f);
  h1[(size_t)p * 64 + c] = fmaxf(m, s * m);
}

// h3[p][c] = mean_d prelu( prelu(h1c_avg, h1lr1) + g2, h1lr2 )
__global__ void k_h3(const float* __restrict__ h1, const float* __restrict__ mapf,
                     const float* __restrict__ h1w, const float* __restrict__ h1b,
                     const float* __restrict__ lr1, const float* __restrict__ lr2,
                     float* __restrict__ h3) {
  int i = blockIdx.x * 256 + threadIdx.x;
  if (i >= P_TOT * GC) return;
  int c = i & 63, p = i >> 6;
  int b = p / HW, hw = p - b * HW;
  int h = hw / 15, w = hw - h * 15;
  float s1 = bslope(lr1[c], 0.999f);
  float s2 = bslope(lr2[c], 0.999f);
  const float* hb = h1 + (size_t)b * HW * 64;
  float acc = 0.f;
  for (int d = 0; d < 4; ++d) {
    float cs = 0.f;
    #pragma unroll
    for (int ii = 0; ii < 11; ++ii) {
      int di = ii - 5;
      int rr = (d == 0) ? 0 : ((d == 3) ? -di : di);
      int cc = (d == 1) ? 0 : di;
      float wgt = h1w[ii * 64 + c];
      int hf_ = h + rr, wf_ = w + cc;
      float fv = ((unsigned)hf_ < 15u && (unsigned)wf_ < 15u)
                     ? hb[(size_t)(hf_ * 15 + wf_) * 64 + c] : 0.f;
      int hbk = h - rr, wbk = w - cc;
      float bv = ((unsigned)hbk < 15u && (unsigned)wbk < 15u)
                     ? hb[(size_t)(hbk * 15 + wbk) * 64 + c] : 0.f;
      cs += 0.5f * wgt * (fv + bv);
    }
    cs += h1b[c];
    float t = fmaxf(cs, s1 * cs) + mapf[(size_t)d * P_TOT * 128 + (size_t)p * 128 + 64 + c];
    t = fmaxf(t, s2 * t);
    acc += t;
  }
  h3[(size_t)p * 64 + c] = 0.25f * acc;
}

// trunk: prelu(h3 + b) -> shuffle -> tc1 -> shuffle -> prelu(tlr1)
__global__ void k_trunk1(const float* __restrict__ h3, const float* __restrict__ h3s,
                         const float* __restrict__ h3b, const float* __restrict__ tc1w,
                         const float* __restrict__ tc1b, const float* __restrict__ tlr1,
                         float* __restrict__ outT) {
  int p = blockIdx.x * 256 + threadIdx.x;
  if (p >= P_TOT) return;
  float t0[64], t1[64], t2[64], t3[64];
  #pragma unroll
  for (int c = 0; c < 64; ++c) {
    float x = h3[(size_t)p * 64 + c] + h3b[c];
    float s = bslope(h3s[c], 0.999f);
    t0[c] = fmaxf(x, s * x);
  }
  #pragma unroll
  for (int gg = 0; gg < 4; ++gg)
    #pragma unroll
    for (int ii = 0; ii < 4; ++ii)
      #pragma unroll
      for (int jj = 0; jj < 4; ++jj)
        t1[gg * 16 + jj * 4 + ii] = t0[gg * 16 + ii * 4 + jj];
  #pragma unroll
  for (int G = 0; G < 16; ++G)
    #pragma unroll
    for (int o = 0; o < 4; ++o) {
      float a = 0.f;
      #pragma unroll
      for (int ii = 0; ii < 4; ++ii) a += t1[G * 4 + ii] * tc1w[(G * 4 + o) * 4 + ii];
      t2[G * 4 + o] = a + tc1b[G * 4 + o];
    }
  #pragma unroll
  for (int gg = 0; gg < 4; ++gg)
    #pragma unroll
    for (int ii = 0; ii < 4; ++ii)
      #pragma unroll
      for (int jj = 0; jj < 4; ++jj)
        t3[gg * 16 + jj * 4 + ii] = t2[gg * 16 + ii * 4 + jj];
  #pragma unroll
  for (int c = 0; c < 64; ++c) {
    float s = bslope(tlr1[c], 0.999f);
    outT[(size_t)p * 64 + c] = fmaxf(t3[c], s * t3[c]);
  }
}

// symmetric 3x3 depthwise conv (pad 1, no bias)
__global__ void k_trunk2(const float* __restrict__ inT, const float* __restrict__ tc2w,
                         float* __restrict__ outT) {
  int i = blockIdx.x * 256 + threadIdx.x;
  if (i >= P_TOT * GC) return;
  int c = i & 63, p = i >> 6;
  int b = p / HW, hw = p - b * HW;
  int h = hw / 15, w = hw - h * 15;
  const float* tb = inT + (size_t)b * HW * 64;
  float w0 = tc2w[c], w1 = tc2w[64 + c], w2 = tc2w[128 + c];
  auto T = [&](int hh, int ww) -> float {
    return ((unsigned)hh < 15u && (unsigned)ww < 15u) ? tb[(size_t)(hh * 15 + ww) * 64 + c] : 0.f;
  };
  float a = w0 * T(h, w);
  a += w1 * (T(h - 1, w) + T(h + 1, w) + T(h, w - 1) + T(h, w + 1));
  a += w2 * (T(h - 1, w - 1) + T(h - 1, w + 1) + T(h + 1, w - 1) + T(h + 1, w + 1));
  outT[(size_t)p * 64 + c] = a;
}

__global__ void k_policy(const float* __restrict__ t2, const float* __restrict__ ps,
                         const float* __restrict__ pb, const float* __restrict__ pw,
                         float* __restrict__ out) {
  int p = blockIdx.x * 256 + threadIdx.x;
  if (p >= P_TOT) return;
  float acc = 0.f;
  #pragma unroll
  for (int c = 0; c < 64; ++c) {
    float x = t2[(size_t)p * 64 + c] + pb[c];
    float s = bslope(ps[c], 0.999f);
    acc += fmaxf(x, s * x) * pw[c];
  }
  out[p] = acc;
}

__global__ void k_vmean(const float* __restrict__ t2, const float* __restrict__ vs,
                        const float* __restrict__ vb, float* __restrict__ vmean) {
  int i = blockIdx.x * 256 + threadIdx.x;
  if (i >= NB * GC) return;
  int c = i & 63, b = i >> 6;
  float s = bslope(vs[c], 0.999f);
  float bias = vb[c];
  float acc = 0.f;
  for (int hw = 0; hw < HW; ++hw) {
    float x = t2[((size_t)b * HW + hw) * 64 + c] + bias;
    acc += fmaxf(x, s * x);
  }
  vmean[b * 64 + c] = acc * (1.f / 225.f);
}

__global__ void k_vmlp(const float* __restrict__ vmean, const float* __restrict__ vlrs,
                       const float* __restrict__ vlrb, const float* __restrict__ v1w,
                       const float* __restrict__ v1b, const float* __restrict__ v2w,
                       const float* __restrict__ v2b, const float* __restrict__ v3w,
                       const float* __restrict__ v3b, const float* __restrict__ vfw,
                       const float* __restrict__ vfb, float* __restrict__ out) {
  __shared__ float sa[64], sb[64];
  int b = blockIdx.x, c = threadIdx.x;
  float x = vmean[b * 64 + c] + vlrb[c];
  float s = bslope(vlrs[c], 10.f);
  sa[c] = fmaxf(x, s * x);
  __syncthreads();
  float a = v1b[c];
  for (int i = 0; i < 64; ++i) a += v1w[c * 64 + i] * sa[i];
  sb[c] = fmaxf(a, 0.f);
  __syncthreads();
  a = v2b[c];
  for (int i = 0; i < 64; ++i) a += v2w[c * 64 + i] * sb[i];
  sa[c] = fmaxf(a, 0.f);
  __syncthreads();
  a = v3b[c];
  for (int i = 0; i < 64; ++i) a += v3w[c * 64 + i] * sa[i];
  sb[c] = fmaxf(a, 0.f);
  __syncthreads();
  if (c < 3) {
    a = vfb[c];
    for (int i = 0; i < 64; ++i) a += vfw[c * 64 + i] * sb[i];
    out[b * 3 + c] = a;
  }
}

// ---------------- launcher ----------------
extern "C" void kernel_launch(void* const* d_in, const int* in_sizes, int n_in,
                              void* d_out, int out_size, void* d_ws, size_t ws_size,
                              hipStream_t stream) {
  (void)in_sizes; (void)n_in; (void)out_size; (void)ws_size;
  const float* x_in  = (const float*)d_in[0];
  const float* fw    = (const float*)d_in[1];
  const float* fb    = (const float*)d_in[2];
  const float* rb_w  = (const float*)d_in[3];
  const float* rb_b  = (const float*)d_in[4];
  const float* rb_w2 = (const float*)d_in[5];
  const float* rb_b2 = (const float*)d_in[6];
  const float* tr_w1 = (const float*)d_in[7];
  const float* tr_b1 = (const float*)d_in[8];
  const float* tr_w2 = (const float*)d_in[9];
  const float* tr_b2 = (const float*)d_in[10];
  const float* fc_w  = (const float*)d_in[11];
  const float* fc_b  = (const float*)d_in[12];
  const float* g1lr  = (const float*)d_in[13];
  const float* h1w   = (const float*)d_in[14];
  const float* h1b   = (const float*)d_in[15];
  const float* h1lr1 = (const float*)d_in[16];
  const float* h1lr2 = (const float*)d_in[17];
  const float* h3lrs = (const float*)d_in[18];
  const float* h3lrb = (const float*)d_in[19];
  const float* tc1w  = (const float*)d_in[20];
  const float* tc1b  = (const float*)d_in[21];
  const float* tlr1  = (const float*)d_in[22];
  const float* tc2w  = (const float*)d_in[23];
  const float* t2ps  = (const float*)d_in[24];
  const float* t2pb  = (const float*)d_in[25];
  const float* t2vs  = (const float*)d_in[26];
  const float* t2vb  = (const float*)d_in[27];
  const float* polw  = (const float*)d_in[28];
  const float* vlrs  = (const float*)d_in[29];
  const float* vlrb  = (const float*)d_in[30];
  const float* v1w   = (const float*)d_in[31];
  const float* v1b   = (const float*)d_in[32];
  const float* v2w   = (const float*)d_in[33];
  const float* v2b   = (const float*)d_in[34];
  const float* v3w   = (const float*)d_in[35];
  const float* v3b   = (const float*)d_in[36];
  const float* vfw   = (const float*)d_in[37];
  const float* vfb   = (const float*)d_in[38];

  char* ws = (char*)d_ws;
  size_t off = 0;
  auto alloc = [&](size_t bytes) -> void* {
    void* p = ws + off;
    off += (bytes + 255) & ~(size_t)255;
    return p;
  };
  u16*   y       = (u16*)  alloc((size_t)4 * P_TOT * 256 * 2);
  u16*   tb      = (u16*)  alloc((size_t)P_TOT * 256 * 2);
  u16*   WlineRb = (u16*)  alloc((size_t)4 * 256 * 768 * 2);
  u16*   Wrb2    = (u16*)  alloc((size_t)4 * 256 * 256 * 2);
  u16*   Wtr1    = (u16*)  alloc((size_t)5 * 256 * 256 * 2);
  u16*   Wtr2    = (u16*)  alloc((size_t)5 * 256 * 256 * 2);
  u16*   Wfc     = (u16*)  alloc((size_t)128 * 256 * 2);
  float* mapf    = (float*)alloc((size_t)4 * P_TOT * 128 * 4);
  float* h1      = (float*)alloc((size_t)P_TOT * 64 * 4);
  float* h3      = (float*)alloc((size_t)P_TOT * 64 * 4);
  float* trunkS  = (float*)alloc((size_t)P_TOT * 64 * 4);
  float* trunk2  = (float*)alloc((size_t)P_TOT * 64 * 4);
  float* vmean   = (float*)alloc((size_t)NB * 64 * 4);

  dim3 blk(256);
  // weight conversion
  k_line_weights<<<(4 * 3 * 256 * 256 + 255) / 256, blk, 0, stream>>>(WlineRb, rb_w);
  k_f32_to_bf16<<<(4 * 65536 + 255) / 256, blk, 0, stream>>>(Wrb2, rb_w2, 4 * 65536);
  k_f32_to_bf16<<<(5 * 65536 + 255) / 256, blk, 0, stream>>>(Wtr1, tr_w1, 5 * 65536);
  k_f32_to_bf16<<<(5 * 65536 + 255) / 256, blk, 0, stream>>>(Wtr2, tr_w2, 5 * 65536);
  k_f32_to_bf16<<<(128 * 256 + 255) / 256, blk, 0, stream>>>(Wfc, fc_w, 128 * 256);

  // initial line conv + swish (all 4 directions)
  k_initial<<<P_TOT, blk, 0, stream>>>(x_in, fw, fb, y);

  dim3 g2(P_TOT / 128, 2);
  // residual blocks: line GEMM (K=768, shift fused) then 1x1 GEMM with residual
  for (int j = 0; j < 4; ++j)
    for (int d = 0; d < 4; ++d) {
      u16* yd = y + (size_t)d * P_TOT * 256;
      k_gemm<0, 1><<<g2, blk, 0, stream>>>(WlineRb + (size_t)j * 256 * 768, yd,
                                           rb_b + j * 256, tb, nullptr, 256, 768, d);
      k_gemm<1, 0><<<g2, blk, 0, stream>>>(Wrb2 + (size_t)j * 65536, tb,
                                           rb_b2 + j * 256, yd, yd, 256, 256, 0);
    }
  // transformer-style 1x1 blocks
  for (int j = 0; j < 5; ++j)
    for (int d = 0; d < 4; ++d) {
      u16* yd = y + (size_t)d * P_TOT * 256;
      k_gemm<0, 0><<<g2, blk, 0, stream>>>(Wtr1 + (size_t)j * 65536, yd,
                                           tr_b1 + j * 256, tb, nullptr, 256, 256, 0);
      k_gemm<1, 0><<<g2, blk, 0, stream>>>(Wtr2 + (size_t)j * 65536, tb,
                                           tr_b2 + j * 256, yd, yd, 256, 256, 0);
    }
  // fc + tanh clamp -> fp32 mapf
  dim3 gfc(P_TOT / 128, 1);
  for (int d = 0; d < 4; ++d)
    k_gemm<2, 0><<<gfc, blk, 0, stream>>>(Wfc, y + (size_t)d * P_TOT * 256, fc_b,
                                          mapf + (size_t)d * P_TOT * 128, nullptr, 128, 256, 0);

  // head
  k_h1<<<(P_TOT * GC + 255) / 256, blk, 0, stream>>>(mapf, g1lr, h1);
  k_h3<<<(P_TOT * GC + 255) / 256, blk, 0, stream>>>(h1, mapf, h1w, h1b, h1lr1, h1lr2, h3);
  k_trunk1<<<(P_TOT + 255) / 256, blk, 0, stream>>>(h3, h3lrs, h3lrb, tc1w, tc1b, tlr1, trunkS);
  k_trunk2<<<(P_TOT * GC + 255) / 256, blk, 0, stream>>>(trunkS, tc2w, trunk2);
  float* out_f = (float*)d_out;
  k_policy<<<(P_TOT + 255) / 256, blk, 0, stream>>>(trunk2, t2ps, t2pb, polw, out_f + NB * 3);
  k_vmean<<<(NB * GC + 255) / 256, blk, 0, stream>>>(trunk2, t2vs, t2vb, vmean);
  k_vmlp<<<NB, 64, 0, stream>>>(vmean, vlrs, vlrb, v1w, v1b, v2w, v2b, v3w, v3b, vfw, vfb, out_f);
}
